// DeepSSM_45732811768504
// MI455X (gfx1250) — compile-verified, matrix operand order checked by reading, not run on hardware
//
#include <hip/hip_runtime.h>

// ---------------------------------------------------------------------------
// Problem constants (from reference)
// ---------------------------------------------------------------------------
#define NL    4
#define DIN   64
#define DMOD  512
#define DST   256
#define BB    8
#define LL    2048
#define MTOT  (BB * LL)      // 16384 tokens
#define CHUNK 128
#define NCHUNK (LL / CHUNK)  // 16

typedef __attribute__((ext_vector_type(16))) __bf16 v16bf;
typedef __attribute__((ext_vector_type(8)))  float  v8f;

union FragAB { uint4 q[2]; v16bf v; };

static __device__ __forceinline__ unsigned short f32_to_bf16(float f) {
    unsigned u = __float_as_uint(f);
    unsigned r = (u + 0x7FFFu + ((u >> 16) & 1u)) >> 16;
    return (unsigned short)r;
}

// gfx1250 async global->LDS copy (16B per lane), tracked by ASYNCcnt.
// VDST VGPR holds the wave-relative LDS byte offset (= low 32 bits of the
// generic pointer for the LDS aperture); VADDR is the 64-bit global address.
static __device__ __forceinline__ void async_copy_b128(const void* gptr, void* lptr) {
    unsigned lds = (unsigned)(unsigned long long)lptr;
    asm volatile("global_load_async_to_lds_b128 %0, %1, off"
                 :: "v"(lds), "v"((unsigned long long)gptr)
                 : "memory");
}

// ---------------------------------------------------------------------------
// Generic bf16 WMMA GEMM:  Cout[M,N] = alpha * A[M,K] * W[N,K]^T (+ Cin)
// A, W row-major bf16 (raw u16). Optional bf16 shadow copy of output.
// block = 256 threads (8 waves), tile = 256(M) x 64(N), K step = 32,
// double-buffered LDS staged with async copies (ASYNCcnt pipelining);
// last K-step peeled so the steady-state loop is branch-free.
// Each wave owns a 32x64 strip = 2(M) x 4(N) WMMA tiles.
// ---------------------------------------------------------------------------
#define LDA 40   // padded LDS stride (shorts): 16 conflict-free b128 rows
#define LDB 40

__launch_bounds__(256)
__global__ void gemm_bf16_wmma(const unsigned short* __restrict__ A,
                               const unsigned short* __restrict__ W,
                               const float* __restrict__ Cin,
                               float* __restrict__ Cout,
                               unsigned short* __restrict__ Cbf,
                               int M, int N, int K, float alpha, int accum)
{
    __shared__ __align__(16) unsigned short As[2][256 * LDA];
    __shared__ __align__(16) unsigned short Bs[2][64 * LDB];

    const int tid  = threadIdx.x;
    const int lane = tid & 31;
    const int wv   = tid >> 5;        // wave 0..7 -> 32-row strip
    const int hf   = lane >> 4;       // half-wave select
    const int l15  = lane & 15;
    const int kb   = hf * 8;          // bf16 A/B fragment: K base per half
    const int mBase = blockIdx.x * 256;
    const int nBase = blockIdx.y * 64;
    const int nk    = K >> 5;         // >= 2 for all shapes used here

    v8f acc[2][4];
#pragma unroll
    for (int mt = 0; mt < 2; ++mt)
#pragma unroll
        for (int nt = 0; nt < 4; ++nt)
#pragma unroll
            for (int r = 0; r < 8; ++r) acc[mt][nt][r] = 0.0f;

    // staging addresses (per-thread, loop-invariant except k0/buf)
    const int aRow = (tid >> 2);            // B-tile row this thread copies
    const int aCol = (tid & 3) * 8;

    // ---- async stage of one K-tile into buffer `buf` (5 copies/thread) ----
    auto stage = [&](int k0, int buf) {
#pragma unroll
        for (int i = 0; i < 4; ++i) {       // A: 256x32 = 1024 16B chunks
            int c  = i * 256 + tid;
            int r  = c >> 2;
            int cc = (c & 3) * 8;
            async_copy_b128(&A[(size_t)(mBase + r) * K + k0 + cc],
                            &As[buf][r * LDA + cc]);
        }
        async_copy_b128(&W[(size_t)(nBase + aRow) * K + k0 + aCol],
                        &Bs[buf][aRow * LDB + aCol]);   // B: 64x32
    };

    // per-wave LDS fragment bases (buffer select resolved by unrolling)
    const int aOff0 = (wv * 32 + l15) * LDA + kb;
    const int aOff1 = aOff0 + 16 * LDA;
    const int bOff  = l15 * LDB + kb;

    auto compute = [&](int p) {
        FragAB af[2], bfr[4];
        const unsigned short* a0 = &As[p][aOff0];
        const unsigned short* a1 = &As[p][aOff1];
        af[0].q[0] = *(const uint4*)a0;
        af[0].q[1] = *(const uint4*)(a0 + 16);
        af[1].q[0] = *(const uint4*)a1;
        af[1].q[1] = *(const uint4*)(a1 + 16);
#pragma unroll
        for (int nt = 0; nt < 4; ++nt) {
            const unsigned short* s = &Bs[p][nt * 16 * LDB + bOff];
            bfr[nt].q[0] = *(const uint4*)s;
            bfr[nt].q[1] = *(const uint4*)(s + 16);
        }
#pragma unroll
        for (int mt = 0; mt < 2; ++mt)
#pragma unroll
            for (int nt = 0; nt < 4; ++nt)
                acc[mt][nt] = __builtin_amdgcn_wmma_f32_16x16x32_bf16(
                    false, af[mt].v, false, bfr[nt].v, (short)0,
                    acc[mt][nt], false, false);
    };

    // ---- pipelined main loop (branch-free body) + peeled last step ----
    stage(0, 0);
    for (int i = 0; i < nk - 1; ++i) {
        stage((i + 1) * 32, (i & 1) ^ 1);                 // prefetch next tile
        asm volatile("s_wait_asynccnt 0x5" ::: "memory"); // my tile-i copies done
        __syncthreads();                                   // everyone's done
        compute(i & 1);
        __syncthreads();      // readers done before this buffer is re-staged
    }
    asm volatile("s_wait_asynccnt 0x0" ::: "memory");
    __syncthreads();
    compute((nk - 1) & 1);

    // Writeback: C/D layout -> row = r + 8*half, col = lane&15 per 16x16 tile
#pragma unroll
    for (int mt = 0; mt < 2; ++mt)
#pragma unroll
        for (int nt = 0; nt < 4; ++nt) {
            int col = nBase + nt * 16 + l15;
#pragma unroll
            for (int r = 0; r < 8; ++r) {
                int row = mBase + wv * 32 + mt * 16 + hf * 8 + r;
                size_t idx = (size_t)row * N + col;
                float v = alpha * acc[mt][nt][r];
                if (accum) v += Cin[idx];
                Cout[idx] = v;
                if (Cbf) Cbf[idx] = f32_to_bf16(v);
            }
        }
}

// ---------------------------------------------------------------------------
// Prep kernels
// ---------------------------------------------------------------------------
__global__ void conv_bf16_kernel(const float* __restrict__ src,
                                 unsigned short* __restrict__ dst, int n)
{
    for (int i = blockIdx.x * blockDim.x + threadIdx.x; i < n;
         i += gridDim.x * blockDim.x)
        dst[i] = f32_to_bf16(src[i]);
}

__global__ void prep_lam_kernel(const float* __restrict__ nu_log,
                                const float* __restrict__ th_log,
                                float* __restrict__ lam,
                                float* __restrict__ lamp,
                                float* __restrict__ gam)
{
    int l = blockIdx.x, n = threadIdx.x;
    int i = l * DST + n;
    float nu = expf(nu_log[i]);
    float th = expf(th_log[i]);
    float r  = expf(-nu);
    float lr = r * cosf(th);
    float li = r * sinf(th);
    gam[i] = sqrtf(fmaxf(0.0f, 1.0f - r * r));
    lam[i * 2] = lr; lam[i * 2 + 1] = li;
    float pr = lr, pi = li;
#pragma unroll
    for (int k = 0; k < 7; ++k) {   // lam^(2^7) = lam^CHUNK
        float nr = pr * pr - pi * pi;
        float ni = 2.0f * pr * pi;
        pr = nr; pi = ni;
    }
    lamp[i * 2] = pr; lamp[i * 2 + 1] = pi;
}

// gamma-fold + convert B (all layers at once)
__global__ void scale_B_kernel(const float* __restrict__ Bre,
                               const float* __restrict__ Bim,
                               const float* __restrict__ gam,
                               unsigned short* __restrict__ oRe,
                               unsigned short* __restrict__ oIm)
{
    int i = blockIdx.x * blockDim.x + threadIdx.x;   // NL*DST*DMOD
    int n = i / DMOD;                                // (layer*DST + n)
    float g = gam[n];
    oRe[i] = f32_to_bf16(Bre[i] * g);
    oIm[i] = f32_to_bf16(Bim[i] * g);
}

// ---------------------------------------------------------------------------
// Chunked parallel scan:  h_t = lam*h_{t-1} + Bu_t   (complex, diagonal)
// ---------------------------------------------------------------------------
__global__ void scan_phase1(const float* __restrict__ bur,
                            const float* __restrict__ bui,
                            const float* __restrict__ lam,
                            float* __restrict__ hr, float* __restrict__ hi,
                            float* __restrict__ carry, int layer)
{
    int n = threadIdx.x;
    int b = blockIdx.x >> 4;
    int c = blockIdx.x & 15;
    int li = (layer * DST + n) * 2;
    float lr = lam[li], lm = lam[li + 1];
    size_t m0 = (size_t)(b * LL + c * CHUNK);
    float ar = 0.0f, ai = 0.0f;
    for (int i = 0; i < CHUNK; ++i) {
        size_t idx = (m0 + i) * DST + n;
        if (i + 16 < CHUNK)
            __builtin_prefetch(&bur[idx + (size_t)16 * DST], 0, 0);
        float br = bur[idx], bi = bui[idx];
        float nr = lr * ar - lm * ai + br;
        float ni = lr * ai + lm * ar + bi;
        ar = nr; ai = ni;
        hr[idx] = ar; hi[idx] = ai;
    }
    int cix = ((b * NCHUNK + c) * DST + n) * 2;
    carry[cix] = ar; carry[cix + 1] = ai;
}

__global__ void scan_phase2(const float* __restrict__ lamp,
                            float* __restrict__ carry, int layer)
{
    int n = threadIdx.x, b = blockIdx.x;
    int li = (layer * DST + n) * 2;
    float pr = lamp[li], pi = lamp[li + 1];
    float Hr = 0.0f, Hi = 0.0f;
    for (int c = 0; c < NCHUNK; ++c) {
        int cix = ((b * NCHUNK + c) * DST + n) * 2;
        float cr = carry[cix], ci = carry[cix + 1];
        carry[cix] = Hr; carry[cix + 1] = Hi;    // incoming state for chunk c
        float nr = pr * Hr - pi * Hi + cr;
        float ni = pr * Hi + pi * Hr + ci;
        Hr = nr; Hi = ni;
    }
}

__global__ void scan_phase3(const float* __restrict__ hr,
                            const float* __restrict__ hi,
                            const float* __restrict__ lam,
                            const float* __restrict__ carry,
                            unsigned short* __restrict__ hbr,
                            unsigned short* __restrict__ hbi, int layer)
{
    int n = threadIdx.x;
    int b = blockIdx.x >> 4;
    int c = blockIdx.x & 15;
    int li = (layer * DST + n) * 2;
    float lr = lam[li], lm = lam[li + 1];
    int cix = ((b * NCHUNK + c) * DST + n) * 2;
    float Hr = carry[cix], Hi = carry[cix + 1];
    float pr = lr, pi = lm;                       // lam^(i+1)
    size_t m0 = (size_t)(b * LL + c * CHUNK);
    for (int i = 0; i < CHUNK; ++i) {
        size_t idx = (m0 + i) * DST + n;
        float fr = pr * Hr - pi * Hi + hr[idx];
        float fi = pr * Hi + pi * Hr + hi[idx];
        hbr[idx] = f32_to_bf16(fr);
        hbi[idx] = f32_to_bf16(fi);
        float nr = pr * lr - pi * lm;
        float ni = pr * lm + pi * lr;
        pr = nr; pi = ni;
    }
}

// ---------------------------------------------------------------------------
// Elementwise: y += D*x ; ybf = bf16(y)
// ---------------------------------------------------------------------------
__global__ void lru_finish_kernel(float* __restrict__ y,
                                  unsigned short* __restrict__ ybf,
                                  const float* __restrict__ x,
                                  const float* __restrict__ Dv, int layer)
{
    size_t i = (size_t)blockIdx.x * blockDim.x + threadIdx.x;
    int d = (int)(i & (DMOD - 1));
    float v = y[i] + Dv[layer * DMOD + d] * x[i];
    y[i] = v;
    ybf[i] = f32_to_bf16(v);
}

// x += sigmoid(res)*z1*sigmoid(z2) ; xbf = bf16(x)
__global__ void glu_res_kernel(float* __restrict__ x,
                               unsigned short* __restrict__ xbf,
                               const float* __restrict__ z1,
                               const float* __restrict__ z2,
                               const float* __restrict__ res, int layer)
{
    size_t i = (size_t)blockIdx.x * blockDim.x + threadIdx.x;
    float s = 1.0f / (1.0f + expf(-res[layer]));
    float g = 1.0f / (1.0f + expf(-z2[i]));
    float nx = x[i] + s * (z1[i] * g);
    x[i] = nx;
    xbf[i] = f32_to_bf16(nx);
}

// ---------------------------------------------------------------------------
// Host orchestration
// ---------------------------------------------------------------------------
extern "C" void kernel_launch(void* const* d_in, const int* in_sizes, int n_in,
                              void* d_out, int out_size, void* d_ws, size_t ws_size,
                              hipStream_t stream)
{
    (void)in_sizes; (void)n_in; (void)out_size; (void)ws_size;
    const float* u_f    = (const float*)d_in[0];
    const float* enc_f  = (const float*)d_in[1];
    const float* dec_f  = (const float*)d_in[2];
    const float* nu_f   = (const float*)d_in[3];
    const float* th_f   = (const float*)d_in[4];
    const float* Bre_f  = (const float*)d_in[5];
    const float* Bim_f  = (const float*)d_in[6];
    const float* Cre_f  = (const float*)d_in[7];
    const float* Cim_f  = (const float*)d_in[8];
    const float* D_f    = (const float*)d_in[9];
    const float* W1_f   = (const float*)d_in[10];
    const float* W2_f   = (const float*)d_in[11];
    const float* res_f  = (const float*)d_in[12];

    char* ws = (char*)d_ws;
    size_t o = 0;
    auto take = [&](size_t bytes) -> char* {
        char* p = ws + o;
        o += (bytes + 255) & ~(size_t)255;
        return p;
    };
    unsigned short* ubf  = (unsigned short*)take((size_t)MTOT * DIN * 2);
    unsigned short* wenc = (unsigned short*)take((size_t)DMOD * DIN * 2);
    unsigned short* wdec = (unsigned short*)take((size_t)DIN * DMOD * 2);
    unsigned short* BreS = (unsigned short*)take((size_t)NL * DST * DMOD * 2);
    unsigned short* BimS = (unsigned short*)take((size_t)NL * DST * DMOD * 2);
    unsigned short* CreB = (unsigned short*)take((size_t)NL * DMOD * DST * 2);
    unsigned short* CimB = (unsigned short*)take((size_t)NL * DMOD * DST * 2);
    unsigned short* W1B  = (unsigned short*)take((size_t)NL * DMOD * DMOD * 2);
    unsigned short* W2B  = (unsigned short*)take((size_t)NL * DMOD * DMOD * 2);
    float* lam   = (float*)take((size_t)NL * DST * 2 * 4);
    float* lamp  = (float*)take((size_t)NL * DST * 2 * 4);
    float* gam   = (float*)take((size_t)NL * DST * 4);
    float* x     = (float*)take((size_t)MTOT * DMOD * 4);
    unsigned short* xbf = (unsigned short*)take((size_t)MTOT * DMOD * 2);
    float* y     = (float*)take((size_t)MTOT * DMOD * 4);
    unsigned short* ybf = (unsigned short*)take((size_t)MTOT * DMOD * 2);
    float* z1    = (float*)take((size_t)MTOT * DMOD * 4);
    float* z2    = (float*)take((size_t)MTOT * DMOD * 4);
    float* bur   = (float*)take((size_t)MTOT * DST * 4);
    float* bui   = (float*)take((size_t)MTOT * DST * 4);
    float* hr    = (float*)take((size_t)MTOT * DST * 4);
    float* hi    = (float*)take((size_t)MTOT * DST * 4);
    unsigned short* hbr = (unsigned short*)take((size_t)MTOT * DST * 2);
    unsigned short* hbi = (unsigned short*)take((size_t)MTOT * DST * 2);
    float* carry = (float*)take((size_t)BB * NCHUNK * DST * 2 * 4);

    auto conv = [&](const float* s, unsigned short* d, int n) {
        conv_bf16_kernel<<<2048, 256, 0, stream>>>(s, d, n);
    };
    auto gemm = [&](const unsigned short* A, const unsigned short* W,
                    const float* Cin, float* Cout, unsigned short* Cbf,
                    int M, int N, int K, float alpha, int accum) {
        dim3 g(M / 256, N / 64);
        gemm_bf16_wmma<<<g, 256, 0, stream>>>(A, W, Cin, Cout, Cbf,
                                              M, N, K, alpha, accum);
    };

    // ---- one-time prep (deterministic, re-run every call) ----
    conv(u_f,   ubf,  MTOT * DIN);
    conv(enc_f, wenc, DMOD * DIN);
    conv(dec_f, wdec, DIN * DMOD);
    conv(Cre_f, CreB, NL * DMOD * DST);
    conv(Cim_f, CimB, NL * DMOD * DST);
    conv(W1_f,  W1B,  NL * DMOD * DMOD);
    conv(W2_f,  W2B,  NL * DMOD * DMOD);
    prep_lam_kernel<<<NL, DST, 0, stream>>>(nu_f, th_f, lam, lamp, gam);
    scale_B_kernel<<<(NL * DST * DMOD) / 256, 256, 0, stream>>>(
        Bre_f, Bim_f, gam, BreS, BimS);

    // ---- encoder: x = u @ enc^T ----
    gemm(ubf, wenc, nullptr, x, xbf, MTOT, DMOD, DIN, 1.0f, 0);

    const int nEl = MTOT * DMOD;
    for (int l = 0; l < NL; ++l) {
        const unsigned short* BreL = BreS + (size_t)l * DST * DMOD;
        const unsigned short* BimL = BimS + (size_t)l * DST * DMOD;
        const unsigned short* CreL = CreB + (size_t)l * DMOD * DST;
        const unsigned short* CimL = CimB + (size_t)l * DMOD * DST;
        const unsigned short* W1L  = W1B  + (size_t)l * DMOD * DMOD;
        const unsigned short* W2L  = W2B  + (size_t)l * DMOD * DMOD;

        // Bu = x @ (gamma*B)^T (complex split)
        gemm(xbf, BreL, nullptr, bur, nullptr, MTOT, DST, DMOD, 1.0f, 0);
        gemm(xbf, BimL, nullptr, bui, nullptr, MTOT, DST, DMOD, 1.0f, 0);

        // chunked scan over L
        scan_phase1<<<BB * NCHUNK, DST, 0, stream>>>(bur, bui, lam, hr, hi, carry, l);
        scan_phase2<<<BB, DST, 0, stream>>>(lamp, carry, l);
        scan_phase3<<<BB * NCHUNK, DST, 0, stream>>>(hr, hi, lam, carry, hbr, hbi, l);

        // y = Re(C h) = h_re@Cre^T - h_im@Cim^T ; then += D*x
        gemm(hbr, CreL, nullptr, y, nullptr, MTOT, DMOD, DST, 1.0f, 0);
        gemm(hbi, CimL, y, y, nullptr, MTOT, DMOD, DST, -1.0f, 1);
        lru_finish_kernel<<<nEl / 256, 256, 0, stream>>>(y, ybf, x, D_f, l);

        // GLU + gated residual
        gemm(ybf, W1L, nullptr, z1, nullptr, MTOT, DMOD, DMOD, 1.0f, 0);
        gemm(ybf, W2L, nullptr, z2, nullptr, MTOT, DMOD, DMOD, 1.0f, 0);
        glu_res_kernel<<<nEl / 256, 256, 0, stream>>>(x, xbf, z1, z2, res_f, l);
    }

    // ---- decoder: out = x @ dec^T ----
    gemm(xbf, wdec, nullptr, (float*)d_out, nullptr, MTOT, DIN, DMOD, 1.0f, 0);
}
